// GatedDeltaNetAttention_71880572666623
// MI455X (gfx1250) — matrix operand order repaired and checked by reading, not compiled, not run on hardware
//
#include <hip/hip_runtime.h>
#include <math.h>

typedef __bf16 bf16_t;
typedef bf16_t v16bf __attribute__((ext_vector_type(16)));
typedef float  v8f   __attribute__((ext_vector_type(8)));
typedef unsigned short ushort8 __attribute__((ext_vector_type(8)));
typedef int    v4i_vec __attribute__((__vector_size__(16)));   // matches async-LDS builtin proto

#define B_DIM 4
#define S_DIM 4096
#define H_DIM 16
#define D_DIM 64
#define HID   1024
#define M_DIM (B_DIM * S_DIM)   // 16384

// Async global->LDS copy path (gfx1250): use only if the toolchain has both
// the async load builtin and the asynccnt wait; otherwise fall back to
// load-to-VGPR + ds_store (still merged into b128 ops by the compiler).
#if defined(__has_builtin)
#  if __has_builtin(__builtin_amdgcn_global_load_async_to_lds_b128) && \
      __has_builtin(__builtin_amdgcn_s_wait_asynccnt)
#    define USE_ASYNC_COPY 1
#  endif
#endif
#ifndef USE_ASYNC_COPY
#  define USE_ASYNC_COPY 0
#endif

__device__ __forceinline__ unsigned short f32_to_bf16(float f) {
  unsigned int u = __builtin_bit_cast(unsigned int, f);
  u += 0x7FFFu + ((u >> 16) & 1u);        // round-to-nearest-even
  return (unsigned short)(u >> 16);
}

__global__ void cvt_f32_to_bf16(const float* __restrict__ in,
                                unsigned short* __restrict__ out, int n) {
  int i = blockIdx.x * blockDim.x + threadIdx.x;
  if (i < n) out[i] = f32_to_bf16(in[i]);
}

// ---------------------------------------------------------------------------
// Tiled bf16 WMMA GEMM: C(MxN, f32) = A(MxK, bf16) * W(KxN, bf16) + bias
// Block tile 128x128, BK=32; 8 waves in 2(M) x 4(N), each wave 64x32 output
// = 4x2 WMMA tiles -> 8 v_wmma_f32_16x16x32_bf16 per K-step per wave.
// FULL=true  : exact tiles, no bounds checks (N multiple of 128)
// FULL=false : column-guarded (used only for the N=16 beta GEMM)
// ACT: 0 = bias only, 1 = bias + sigmoid
// ---------------------------------------------------------------------------
#define BM 128
#define BN 128
#define BK 32
#define LDT 40   // BK + 8 pad (ushorts); 80 B row stride keeps 16-B alignment

template<bool FULL, int ACT>
__global__ __launch_bounds__(256)
void gemm_bf16_wmma(const unsigned short* __restrict__ A,   // M x K bf16
                    const unsigned short* __restrict__ W,   // K x N bf16
                    const float* __restrict__ bias,         // N
                    float* __restrict__ C,                  // M x N f32
                    int K, int N) {
  __shared__ __align__(16) unsigned short sX[BM * LDT];
  __shared__ __align__(16) unsigned short sW[BN * LDT];

  const int tid  = threadIdx.x;
  const int m0   = blockIdx.x * BM;
  const int n0   = blockIdx.y * BN;
  const int wave = tid >> 5;
  const int lane = tid & 31;
  const int half = lane >> 4;         // 16-lane half of the wave
  const int l16  = lane & 15;
  const int wm   = (wave & 1) * 64;   // 2 waves along M
  const int wn   = (wave >> 1) * 32;  // 4 waves along N

  v8f acc[4][2];
  #pragma unroll
  for (int a = 0; a < 4; ++a)
    #pragma unroll
    for (int b = 0; b < 2; ++b)
      #pragma unroll
      for (int e = 0; e < 8; ++e) acc[a][b][e] = 0.0f;

  // A-tile thread mapping: two 16-B segments per thread
  const int asg_m  = (tid * 2) >> 2;          // row of first segment
  const int asg_k0 = ((tid * 2) & 3) * 8;     // k-offset of first segment
  // W-tile thread mapping: k = tid>>3, 16 consecutive n per thread
  const int wk   = tid >> 3;                  // 0..31
  const int wn16 = (tid & 7) * 16;            // 0..112

  for (int kb = 0; kb < K; kb += BK) {
    // ---- A tile (row-major 128x32): async copy straight into LDS
    #pragma unroll
    for (int i = 0; i < 2; ++i) {
      int seg = tid * 2 + i;
      int m   = seg >> 2;
      int ks  = (seg & 3) * 8;
      const unsigned short* gsrc = A + (size_t)(m0 + m) * K + kb + ks;
      unsigned short* ldst = &sX[m * LDT + ks];
#if USE_ASYNC_COPY
      __builtin_amdgcn_global_load_async_to_lds_b128(
          (v4i_vec*)gsrc, (v4i_vec*)ldst, 0, 0);
#else
      *(ushort8*)ldst = *(const ushort8*)gsrc;
#endif
    }
    // ---- W tile TRANSPOSED into sW[n][k] so each lane's K-run is contiguous
    if (FULL) {
      ushort8 w0 = *(const ushort8*)(W + (size_t)(kb + wk) * N + n0 + wn16);
      ushort8 w1 = *(const ushort8*)(W + (size_t)(kb + wk) * N + n0 + wn16 + 8);
      #pragma unroll
      for (int j = 0; j < 8; ++j) {
        sW[(wn16 + j) * LDT + wk]     = w0[j];
        sW[(wn16 + 8 + j) * LDT + wk] = w1[j];
      }
    } else {
      #pragma unroll
      for (int j = 0; j < 16; ++j) {
        int n = n0 + wn16 + j;
        sW[(wn16 + j) * LDT + wk] =
            (n < N) ? W[(size_t)(kb + wk) * N + n] : (unsigned short)0;
      }
    }
    // ---- prefetch next K-tile (global_prefetch_b8)
    if (kb + BK < K) {
      __builtin_prefetch(A + (size_t)(m0 + asg_m) * K + kb + BK + asg_k0, 0, 3);
      __builtin_prefetch(W + (size_t)(kb + BK + wk) * N + n0 + wn16, 0, 3);
    }
#if USE_ASYNC_COPY
    __builtin_amdgcn_s_wait_asynccnt(0);
#endif
    __syncthreads();

    // ---- fragments per CDNA5 wave32 WMMA layouts
    union Frag { v16bf v; ushort8 h[2]; };
    Frag af[4], bfr[2];
    #pragma unroll
    for (int a = 0; a < 4; ++a) {
      int row = wm + a * 16 + l16;   // A: lane = row m, elems0-7 K=half*8+j, elems8-15 K=16+half*8+j
      af[a].h[0] = *(const ushort8*)&sX[row * LDT + half * 8];
      af[a].h[1] = *(const ushort8*)&sX[row * LDT + 16 + half * 8];
    }
    #pragma unroll
    for (int b = 0; b < 2; ++b) {
      int col = wn + b * 16 + l16;   // B: lane = col n, elems0-7 K=half*8+j, elems8-15 K=16+half*8+j
      bfr[b].h[0] = *(const ushort8*)&sW[col * LDT + half * 8];
      bfr[b].h[1] = *(const ushort8*)&sW[col * LDT + 16 + half * 8];
    }
    #pragma unroll
    for (int a = 0; a < 4; ++a)
      #pragma unroll
      for (int b = 0; b < 2; ++b)
        acc[a][b] = __builtin_amdgcn_wmma_f32_16x16x32_bf16(
            false, af[a].v, false, bfr[b].v, (short)0, acc[a][b], false, false);
    __syncthreads();
  }

  // ---- epilogue: D layout -> element i is row half*8+i, col l16
  #pragma unroll
  for (int b = 0; b < 2; ++b) {
    int col = n0 + wn + b * 16 + l16;
    if (!FULL && col >= N) continue;
    float bv = bias[col];
    #pragma unroll
    for (int a = 0; a < 4; ++a) {
      int rbase = m0 + wm + a * 16 + half * 8;
      #pragma unroll
      for (int i = 0; i < 8; ++i) {
        float r = acc[a][b][i] + bv;
        if (ACT == 1) r = 1.0f / (1.0f + __expf(-r));   // sigmoid for beta
        C[(size_t)(rbase + i) * N + col] = r;
      }
    }
  }
}

// ---------------------------------------------------------------------------
// k <- k / (||k||_2 + 1e-6) per (b,s,h) 64-vector; one vector per wave32
// ---------------------------------------------------------------------------
__global__ __launch_bounds__(256)
void normalize_k(float* __restrict__ k, int nvec) {
  int wave = threadIdx.x >> 5;
  int lane = threadIdx.x & 31;
  int vec  = blockIdx.x * 8 + wave;
  if (vec >= nvec) return;
  float2* p = (float2*)(k + (size_t)vec * 64 + lane * 2);
  float2 kv = *p;
  float s = kv.x * kv.x + kv.y * kv.y;
  #pragma unroll
  for (int off = 16; off > 0; off >>= 1) s += __shfl_xor(s, off, 32);
  float inv = 1.0f / (sqrtf(s) + 1e-6f);
  kv.x *= inv; kv.y *= inv;
  *p = kv;
}

// ---------------------------------------------------------------------------
// Sequential delta-rule scan. One block per (b,h); thread d owns state row
// S[d][0..63] in VGPRs. k/q/v/beta broadcast through LDS each timestep.
// Output written as bf16 directly for the Wo WMMA GEMM.
// ---------------------------------------------------------------------------
__global__ __launch_bounds__(64)
void deltanet_scan(const float* __restrict__ q, const float* __restrict__ k,
                   const float* __restrict__ v, const float* __restrict__ beta,
                   unsigned short* __restrict__ attn) {
  const int b = blockIdx.x / H_DIM;
  const int h = blockIdx.x % H_DIM;
  const int d = threadIdx.x;            // 0..63: state row

  float Srow[D_DIM];
  #pragma unroll
  for (int e = 0; e < D_DIM; ++e) Srow[e] = 0.0f;

  __shared__ float lk[D_DIM], lq[D_DIM], lv[D_DIM];
  __shared__ float lbeta;

  for (int t = 0; t < S_DIM; ++t) {
    size_t base = ((size_t)(b * S_DIM + t)) * HID + h * D_DIM;
    lk[d] = k[base + d];
    lq[d] = q[base + d];
    lv[d] = v[base + d];
    if (d == 0) lbeta = beta[(size_t)(b * S_DIM + t) * H_DIM + h];
    __syncthreads();

    float sk = 0.0f;
    #pragma unroll
    for (int e = 0; e < D_DIM; ++e) sk += Srow[e] * lk[e];
    float bd = lbeta * (lv[d] - sk);
    float o = 0.0f;
    #pragma unroll
    for (int e = 0; e < D_DIM; ++e) {
      Srow[e] += bd * lk[e];            // rank-1 update
      o += Srow[e] * lq[e];             // o_t = S_new @ q_t
    }
    attn[base + d] = f32_to_bf16(o);
    __syncthreads();
  }
}

// ---------------------------------------------------------------------------
extern "C" void kernel_launch(void* const* d_in, const int* in_sizes, int n_in,
                              void* d_out, int out_size, void* d_ws, size_t ws_size,
                              hipStream_t stream) {
  (void)in_sizes; (void)n_in; (void)out_size; (void)ws_size;
  const float* x     = (const float*)d_in[0];
  const float* Wq    = (const float*)d_in[1];
  const float* bq    = (const float*)d_in[2];
  const float* Wk    = (const float*)d_in[3];
  const float* bk    = (const float*)d_in[4];
  const float* Wv    = (const float*)d_in[5];
  const float* bv    = (const float*)d_in[6];
  const float* Wbeta = (const float*)d_in[7];
  const float* bbeta = (const float*)d_in[8];
  const float* Wo    = (const float*)d_in[9];
  const float* bo    = (const float*)d_in[10];

  char* ws = (char*)d_ws;
  auto alloc = [&](size_t bytes) -> void* {
    void* p = (void*)ws;
    ws += (bytes + 255) & ~(size_t)255;
    return p;
  };

  const size_t M = M_DIM, K = HID;
  unsigned short* xb    = (unsigned short*)alloc(M * K * 2);
  unsigned short* wqb   = (unsigned short*)alloc(K * HID * 2);
  unsigned short* wkb   = (unsigned short*)alloc(K * HID * 2);
  unsigned short* wvb   = (unsigned short*)alloc(K * HID * 2);
  unsigned short* wbb   = (unsigned short*)alloc(K * H_DIM * 2);
  unsigned short* wob   = (unsigned short*)alloc(K * HID * 2);
  float*          qf    = (float*)alloc(M * HID * 4);
  float*          kf    = (float*)alloc(M * HID * 4);
  float*          vf    = (float*)alloc(M * HID * 4);
  float*          betaf = (float*)alloc(M * H_DIM * 4);
  unsigned short* attnb = (unsigned short*)alloc(M * HID * 2);

  auto cvt = [&](const float* src, unsigned short* dst, size_t n) {
    cvt_f32_to_bf16<<<(unsigned)((n + 255) / 256), 256, 0, stream>>>(src, dst, (int)n);
  };
  cvt(x,     xb,  M * K);
  cvt(Wq,    wqb, K * HID);
  cvt(Wk,    wkb, K * HID);
  cvt(Wv,    wvb, K * HID);
  cvt(Wbeta, wbb, K * H_DIM);
  cvt(Wo,    wob, K * HID);

  dim3 blk(256);
  dim3 grd_big(M_DIM / BM, HID / BN);   // (128, 8)
  gemm_bf16_wmma<true, 0><<<grd_big, blk, 0, stream>>>(xb, wqb, bq, qf, HID, HID);
  gemm_bf16_wmma<true, 0><<<grd_big, blk, 0, stream>>>(xb, wkb, bk, kf, HID, HID);
  gemm_bf16_wmma<true, 0><<<grd_big, blk, 0, stream>>>(xb, wvb, bv, vf, HID, HID);
  dim3 grd_beta(M_DIM / BM, 1);
  gemm_bf16_wmma<false, 1><<<grd_beta, blk, 0, stream>>>(xb, wbb, bbeta, betaf, HID, H_DIM);

  int nvec = B_DIM * S_DIM * H_DIM;             // 262144 head-vectors
  normalize_k<<<nvec / 8, 256, 0, stream>>>(kf, nvec);

  deltanet_scan<<<B_DIM * H_DIM, D_DIM, 0, stream>>>(qf, kf, vf, betaf, attnb);

  gemm_bf16_wmma<true, 0><<<grd_big, blk, 0, stream>>>(attnb, wob, bo, (float*)d_out, HID, HID);
}
